// DecoderAtten_20581483282909
// MI455X (gfx1250) — matrix operand order repaired and checked by reading, not compiled, not run on hardware
//
#include <hip/hip_runtime.h>
#include <hip/hip_bf16.h>

// ---------------- problem dimensions ----------------
#define B_    128
#define S_    100
#define EMB_  256
#define ENC2  1024   // 2*ENC
#define DEC_  512
#define V_    32000
#define G3    1536   // 3*DEC
#define XK    1280   // EMB + 2*ENC   (GRU input width)
#define OUTK  1792   // DEC + 2*ENC + EMB (output proj K)
#define NEGV  (-1e10f)

typedef __attribute__((ext_vector_type(16))) __bf16 v16bf;
typedef __attribute__((ext_vector_type(8)))  __bf16 v8bf;
typedef __attribute__((ext_vector_type(4)))  __bf16 v4bf;
typedef __attribute__((ext_vector_type(2)))  __bf16 v2bf;
typedef __attribute__((ext_vector_type(8)))  float  v8f;
typedef __attribute__((ext_vector_type(4)))  float  v4f;
typedef __attribute__((ext_vector_type(2)))  float  v2f;

// ---------------- WMMA helpers ----------------
static __device__ __forceinline__ v8f wmma_bf16(v16bf a, v16bf b, v8f c) {
    // D = A(16x32 bf16) x B(32x16 bf16) + C(16x16 f32)
    return __builtin_amdgcn_wmma_f32_16x16x32_bf16(
        /*neg_a=*/false, a, /*neg_b=*/false, b,
        /*c_mod=*/(short)0, c, /*reuse_a=*/false, /*reuse_b=*/false);
}

static __device__ __forceinline__ v4bf cvt4(v4f x) {
    v4bf r; r[0] = (__bf16)x[0]; r[1] = (__bf16)x[1]; r[2] = (__bf16)x[2]; r[3] = (__bf16)x[3];
    return r;
}

// A-fragment (16x32, 16-bit) per ISA layout:
// lane L: m = L&15, half h = L>>4; VGPR v<4 -> k = h*8+2v ; v>=4 -> k = 16+h*8+2(v-4)
static __device__ __forceinline__ v16bf load_a_f32(const float* src, int ld, int lane) {
    v16bf a;
    int m = lane & 15, h = lane >> 4;
#pragma unroll
    for (int v = 0; v < 8; ++v) {
        int k = (v < 4) ? (h * 8 + v * 2) : (16 + h * 8 + (v - 4) * 2);
        v2f t = *(const v2f*)(src + (size_t)m * ld + k);   // 8B-aligned (k even)
        a[2 * v]     = (__bf16)t[0];
        a[2 * v + 1] = (__bf16)t[1];
    }
    return a;
}
static __device__ __forceinline__ v16bf load_a_lds(const __bf16* src, int ld, int lane) {
    v16bf a;
    int m = lane & 15, h = lane >> 4;
#pragma unroll
    for (int v = 0; v < 8; ++v) {
        int k = (v < 4) ? (h * 8 + v * 2) : (16 + h * 8 + (v - 4) * 2);
        v2bf t = *(const v2bf*)(src + (size_t)m * ld + k); // 4B-aligned pair
        a[2 * v]     = t[0];
        a[2 * v + 1] = t[1];
    }
    return a;
}
// B-fragment (32x16, 16-bit): lane L: n = L&15, h = L>>4; elem s<8 -> k = h*8+s ; s>=8 -> k = 16+h*8+(s-8)
static __device__ __forceinline__ v16bf load_b_f32(const float* src, int ld, int lane) {
    v16bf b;
    int n = lane & 15, h = lane >> 4;
#pragma unroll
    for (int s = 0; s < 16; ++s) {
        int k = (s < 8) ? (h * 8 + s) : (16 + h * 8 + (s - 8));
        b[s] = (__bf16)src[(size_t)k * ld + n];
    }
    return b;
}
// non-temporal variant for streamed weights (W_out: 229MB > 192MB L2, read once)
static __device__ __forceinline__ v16bf load_b_f32_nt(const float* src, int ld, int lane) {
    v16bf b;
    int n = lane & 15, h = lane >> 4;
#pragma unroll
    for (int s = 0; s < 16; ++s) {
        int k = (s < 8) ? (h * 8 + s) : (16 + h * 8 + (s - 8));
        b[s] = (__bf16)__builtin_nontemporal_load(src + (size_t)k * ld + n);
    }
    return b;
}
static __device__ __forceinline__ v16bf load_b_bf16(const __bf16* src, int ld, int lane) {
    v16bf b;
    int n = lane & 15, h = lane >> 4;
#pragma unroll
    for (int s = 0; s < 16; ++s) {
        int k = (s < 8) ? (h * 8 + s) : (16 + h * 8 + (s - 8));
        b[s] = src[(size_t)k * ld + n];
    }
    return b;
}
// B-fragment from a row-major [N,K] weight (i.e. B = W^T): lane L: n = L&15, k per A table order; elem = W[n*ldk + k]
static __device__ __forceinline__ v16bf load_bt_f32(const float* Wbase, int ldk, int lane) {
    v16bf b;
    int n = lane & 15, h = lane >> 4;
#pragma unroll
    for (int v = 0; v < 8; ++v) {
        int k = (v < 4) ? (h * 8 + v * 2) : (16 + h * 8 + (v - 4) * 2);
        v2f t = *(const v2f*)(Wbase + (size_t)n * ldk + k);
        b[2 * v]     = (__bf16)t[0];
        b[2 * v + 1] = (__bf16)t[1];
    }
    return b;
}

static __device__ __forceinline__ float sigmoidf_(float x) { return 1.f / (1.f + __expf(-x)); }

// ---------------- kernels ----------------

// 0) W_e (rows DEC..DEC+ENC2 of W_attn) -> bf16, row-major [1024,512]
__global__ __launch_bounds__(256) void k_we_convert(const float* __restrict__ W_attn,
                                                    __bf16* __restrict__ We) {
    int i4 = blockIdx.x * 256 + threadIdx.x;           // 131072 quads
    v4f x = *(const v4f*)(W_attn + (size_t)DEC_ * DEC_ + (size_t)i4 * 4);
    *(v4bf*)(We + (size_t)i4 * 4) = cvt4(x);
}

// 1) hp = hidden @ W_h + b_attn    [128,512]
__global__ __launch_bounds__(256) void k_hproj(const float* __restrict__ hidden,
                                               const float* __restrict__ W_attn,
                                               const float* __restrict__ b_attn,
                                               float* __restrict__ hp) {
    int wave = blockIdx.x * (blockDim.x >> 5) + (threadIdx.x >> 5); // 256 waves
    int lane = threadIdx.x & 31;
    int mt = wave >> 5;       // 0..7
    int nt = wave & 31;       // 0..31
    v8f acc = {};
    for (int kc = 0; kc < DEC_ / 32; ++kc) {
        v16bf a = load_a_f32(hidden + (size_t)(mt * 16) * DEC_ + kc * 32, DEC_, lane);
        v16bf b = load_b_f32(W_attn + (size_t)(kc * 32) * DEC_ + nt * 16, DEC_, lane);
        acc = wmma_bf16(a, b, acc);
    }
    int nl = lane & 15, h = lane >> 4;
    int n = nt * 16 + nl;
    float bias = b_attn[n];
#pragma unroll
    for (int r = 0; r < 8; ++r) {
        int m = mt * 16 + r + h * 8;
        hp[(size_t)m * DEC_ + n] = acc[r] + bias;
    }
}

// 2) scores[b][s] = v_attn . tanh(hp[b] + enc[s,b] @ W_e)   (masked)
__global__ __launch_bounds__(256) void k_scores(const float* __restrict__ enc,
                                                const __bf16* __restrict__ We,
                                                const float* __restrict__ hp,
                                                const float* __restrict__ v_attn,
                                                const int* __restrict__ mask,
                                                float* __restrict__ scores) {
    __shared__ __bf16 As[16 * ENC2];   // 32 KB A tile
    __shared__ float  srow[16];
    int tid = threadIdx.x;
    int t = blockIdx.x;                // 0..799
    int s = t >> 3;
    int b0 = (t & 7) * 16;
    // stage + convert with b128 loads (16 quads per thread)
    for (int i = tid; i < (16 * ENC2) / 4; i += 256) {
        int row = i >> 8;              // i*4 / 1024
        int col = (i & 255) * 4;
        v4f x = *(const v4f*)(enc + ((size_t)s * B_ + b0 + row) * ENC2 + col);
        *(v4bf*)(As + row * ENC2 + col) = cvt4(x);
    }
    if (tid < 16) srow[tid] = 0.f;
    __syncthreads();

    int wave = tid >> 5, lane = tid & 31;
    int nl = lane & 15, h = lane >> 4;
#pragma unroll 1
    for (int q = 0; q < 4; ++q) {
        int nt = wave * 4 + q;         // 0..31
        v8f acc = {};
        for (int kc = 0; kc < ENC2 / 32; ++kc) {
            v16bf a = load_a_lds(As + kc * 32, ENC2, lane);
            v16bf b = load_b_bf16(We + (size_t)(kc * 32) * DEC_ + nt * 16, DEC_, lane);
            acc = wmma_bf16(a, b, acc);
        }
        int n = nt * 16 + nl;
        float vn = v_attn[n];
#pragma unroll
        for (int r = 0; r < 8; ++r) {
            int brow = b0 + r + h * 8;
            float e = tanhf(acc[r] + hp[(size_t)brow * DEC_ + n]) * vn;
            for (int off = 8; off; off >>= 1) e += __shfl_xor(e, off, 32);
            if (nl == 0) atomicAdd(&srow[r + h * 8], e);
        }
    }
    __syncthreads();
    if (tid < 16) {
        int b = b0 + tid;
        float sc = srow[tid];
        scores[b * S_ + s] = (mask[b * S_ + s] == 0) ? NEGV : sc;
    }
}

// 3) softmax over S per row b -> a  (written straight into d_out region 3)
__global__ __launch_bounds__(128) void k_softmax(const float* __restrict__ scores,
                                                 float* __restrict__ a_out) {
    __shared__ float red[128];
    int b = blockIdx.x, tid = threadIdx.x;
    float x = (tid < S_) ? scores[b * S_ + tid] : -3.4e38f;
    red[tid] = x;
    __syncthreads();
    for (int off = 64; off; off >>= 1) {
        if (tid < off) red[tid] = fmaxf(red[tid], red[tid + off]);
        __syncthreads();
    }
    float mx = red[0];
    __syncthreads();
    float ex = (tid < S_) ? __expf(x - mx) : 0.f;
    red[tid] = ex;
    __syncthreads();
    for (int off = 64; off; off >>= 1) {
        if (tid < off) red[tid] += red[tid + off];
        __syncthreads();
    }
    float inv = 1.f / red[0];
    if (tid < S_) a_out[b * S_ + tid] = ex * inv;
}

// 4) weighted[b][e] = sum_s a[b][s] * enc[s][b][e]
//    a-row staged into LDS with CDNA5 async global->LDS copy (ASYNCcnt path)
__global__ __launch_bounds__(256) void k_weighted(const float* __restrict__ a,
                                                  const float* __restrict__ enc,
                                                  float* __restrict__ weighted) {
    __shared__ float ash[S_];
    int blk = blockIdx.x;              // 4 blocks per b
    int b = blk >> 2;
    int e = (blk & 3) * 256 + threadIdx.x;
    if (threadIdx.x < S_) {
        unsigned     lds_off = (unsigned)(uintptr_t)&ash[threadIdx.x]; // low 32b = LDS offset
        const float* gsrc    = a + b * S_ + threadIdx.x;
        asm volatile("global_load_async_to_lds_b32 %0, %1, off"
                     :: "v"(lds_off), "v"(gsrc)
                     : "memory");
    }
    asm volatile("s_wait_asynccnt 0x0" ::: "memory");
    __syncthreads();
    float acc = 0.f;
#pragma unroll 4
    for (int s = 0; s < S_; ++s)
        acc += ash[s] * enc[((size_t)s * B_ + b) * ENC2 + e];
    weighted[(size_t)b * ENC2 + e] = acc;
}

// 5) embedding gather
__global__ __launch_bounds__(256) void k_embed(const int* __restrict__ tokens,
                                               const float* __restrict__ table,
                                               float* __restrict__ embedded) {
    int i = blockIdx.x * 256 + threadIdx.x;  // 128*256
    int b = i >> 8, e = i & 255;
    embedded[i] = table[(size_t)tokens[b] * EMB_ + e];
}

// 6) gx = [embedded; weighted] @ W_ih^T + b_ih    [128,1536], K = 1280
__global__ __launch_bounds__(256) void k_gx(const float* __restrict__ emb,
                                            const float* __restrict__ wgt,
                                            const float* __restrict__ W_ih,
                                            const float* __restrict__ b_ih,
                                            float* __restrict__ gx) {
    int mt = blockIdx.y;                       // 0..7
    int wave = threadIdx.x >> 5, lane = threadIdx.x & 31;
    int nt = blockIdx.x * 8 + wave;            // 0..95
    int mrow = lane & 15, h = lane >> 4;
    v8f acc = {};
    for (int kc = 0; kc < XK / 32; ++kc) {
        v16bf a;
#pragma unroll
        for (int v = 0; v < 8; ++v) {
            int k = kc * 32 + ((v < 4) ? (h * 8 + v * 2) : (16 + h * 8 + (v - 4) * 2));
            int m = mt * 16 + mrow;
            // region boundary (256) is even: a (k,k+1) pair never straddles it
            const float* p = (k < EMB_) ? emb + (size_t)m * EMB_ + k
                                        : wgt + (size_t)m * ENC2 + (k - EMB_);
            v2f tp = *(const v2f*)p;
            a[2 * v]     = (__bf16)tp[0];
            a[2 * v + 1] = (__bf16)tp[1];
        }
        v16bf b = load_bt_f32(W_ih + (size_t)(nt * 16) * XK + kc * 32, XK, lane);
        acc = wmma_bf16(a, b, acc);
    }
    int nl = lane & 15;
    int n = nt * 16 + nl;
    float bias = b_ih[n];
#pragma unroll
    for (int r = 0; r < 8; ++r) {
        int m = mt * 16 + r + 8 * h;
        gx[(size_t)m * G3 + n] = acc[r] + bias;
    }
}

// 7) gh = hidden @ W_hh^T + b_hh    [128,1536], K = 512
__global__ __launch_bounds__(256) void k_gh(const float* __restrict__ hidden,
                                            const float* __restrict__ W_hh,
                                            const float* __restrict__ b_hh,
                                            float* __restrict__ gh) {
    int mt = blockIdx.y;
    int wave = threadIdx.x >> 5, lane = threadIdx.x & 31;
    int nt = blockIdx.x * 8 + wave;
    v8f acc = {};
    for (int kc = 0; kc < DEC_ / 32; ++kc) {
        v16bf a = load_a_f32(hidden + (size_t)(mt * 16) * DEC_ + kc * 32, DEC_, lane);
        v16bf b = load_bt_f32(W_hh + (size_t)(nt * 16) * DEC_ + kc * 32, DEC_, lane);
        acc = wmma_bf16(a, b, acc);
    }
    int nl = lane & 15, h = lane >> 4;
    int n = nt * 16 + nl;
    float bias = b_hh[n];
#pragma unroll
    for (int r = 0; r < 8; ++r) {
        int m = mt * 16 + r + 8 * h;
        gh[(size_t)m * G3 + n] = acc[r] + bias;
    }
}

// 8) GRU gate fusion -> h_new (d_out region 2)
__global__ __launch_bounds__(256) void k_gru(const float* __restrict__ gx,
                                             const float* __restrict__ gh,
                                             const float* __restrict__ hidden,
                                             float* __restrict__ h_new) {
    int i = blockIdx.x * 256 + threadIdx.x;   // 128*512
    int b = i >> 9, d = i & 511;
    size_t o = (size_t)b * G3 + d;
    float r = sigmoidf_(gx[o] + gh[o]);
    float z = sigmoidf_(gx[o + 512] + gh[o + 512]);
    float n = tanhf(gx[o + 1024] + r * gh[o + 1024]);
    h_new[i] = (1.f - z) * n + z * hidden[i];
}

// 9) logits = [h_new; weighted; embedded] @ W_out + b_out   [128, 32000]
//    memory-bound: W_out streamed once with NT b32 loads; NT stores for logits.
__global__ __launch_bounds__(256) void k_logits(const float* __restrict__ h_new,
                                                const float* __restrict__ wgt,
                                                const float* __restrict__ emb,
                                                const float* __restrict__ W_out,
                                                const float* __restrict__ b_out,
                                                float* __restrict__ logits) {
    __shared__ __bf16 As[128 * 32];   // 8 KB A chunk: 128 rows x 32 k
    int tid = threadIdx.x;
    int wave = tid >> 5, lane = tid & 31;
    int colbase = blockIdx.x * 128 + wave * 16;
    v8f acc[8];
#pragma unroll
    for (int mt = 0; mt < 8; ++mt)
#pragma unroll
        for (int r = 0; r < 8; ++r) acc[mt][r] = 0.f;

    for (int kc = 0; kc < OUTK / 32; ++kc) {
        // chunk boundaries (512, 1536) are multiples of 32 -> uniform source per kc
        int kg = kc * 32;
        const float* src; int ld, k0;
        if (kg < DEC_)             { src = h_new; ld = DEC_; k0 = kg; }
        else if (kg < DEC_ + ENC2) { src = wgt;   ld = ENC2; k0 = kg - DEC_; }
        else                       { src = emb;   ld = EMB_; k0 = kg - DEC_ - ENC2; }
        for (int i = tid; i < 128 * 8; i += 256) {   // 8 quads per row
            int m = i >> 3, kk = (i & 7) * 4;
            v4f x = *(const v4f*)(src + (size_t)m * ld + k0 + kk);
            *(v4bf*)(As + m * 32 + kk) = cvt4(x);
        }
        __syncthreads();
        v16bf b = load_b_f32_nt(W_out + (size_t)kg * V_ + colbase, V_, lane);
#pragma unroll
        for (int mt = 0; mt < 8; ++mt) {
            v16bf a = load_a_lds(As + mt * 16 * 32, 32, lane);
            acc[mt] = wmma_bf16(a, b, acc[mt]);
        }
        __syncthreads();
    }
    int nl = lane & 15, h = lane >> 4;
    int col = colbase + nl;
    float bias = b_out[col];
#pragma unroll
    for (int mt = 0; mt < 8; ++mt)
#pragma unroll
        for (int r = 0; r < 8; ++r) {
            int m = mt * 16 + r + 8 * h;
            __builtin_nontemporal_store(acc[mt][r] + bias, &logits[(size_t)m * V_ + col]);
        }
}

// ---------------- launcher ----------------
extern "C" void kernel_launch(void* const* d_in, const int* in_sizes, int n_in,
                              void* d_out, int out_size, void* d_ws, size_t ws_size,
                              hipStream_t stream) {
    const int*   tokens   = (const int*)  d_in[0];
    const float* hidden   = (const float*)d_in[1];
    const float* enc      = (const float*)d_in[2];
    const int*   mask     = (const int*)  d_in[3];
    const float* emb_tab  = (const float*)d_in[4];
    const float* W_attn   = (const float*)d_in[5];
    const float* b_attn   = (const float*)d_in[6];
    const float* v_attn   = (const float*)d_in[7];
    const float* W_ih     = (const float*)d_in[8];
    const float* W_hh     = (const float*)d_in[9];
    const float* b_ih     = (const float*)d_in[10];
    const float* b_hh     = (const float*)d_in[11];
    const float* W_out    = (const float*)d_in[12];
    const float* b_out    = (const float*)d_in[13];

    float* out    = (float*)d_out;
    float* logits = out;                               // [128, 32000]
    float* h_new  = out + (size_t)B_ * V_;             // [128, 512]
    float* a_out  = h_new + (size_t)B_ * DEC_;         // [128, 100]

    char* w = (char*)d_ws;
    float*  hp       = (float*) w;  w += (size_t)B_ * DEC_ * 4;      // 256 KB
    __bf16* We       = (__bf16*)w;  w += (size_t)ENC2 * DEC_ * 2;    // 1 MB
    float*  scores   = (float*) w;  w += (size_t)B_ * S_ * 4;        // 50 KB
    float*  weighted = (float*) w;  w += (size_t)B_ * ENC2 * 4;      // 512 KB
    float*  embedded = (float*) w;  w += (size_t)B_ * EMB_ * 4;      // 128 KB
    float*  gx       = (float*) w;  w += (size_t)B_ * G3 * 4;        // 768 KB
    float*  gh       = (float*) w;  w += (size_t)B_ * G3 * 4;        // 768 KB

    k_we_convert<<<(ENC2 * DEC_) / (256 * 4), 256, 0, stream>>>(W_attn, We);
    k_hproj     <<<32, 256, 0, stream>>>(hidden, W_attn, b_attn, hp);
    k_scores    <<<(S_ * B_) / 16, 256, 0, stream>>>(enc, We, hp, v_attn, mask, scores);
    k_softmax   <<<B_, 128, 0, stream>>>(scores, a_out);
    k_weighted  <<<(B_ * ENC2) / 256, 256, 0, stream>>>(a_out, enc, weighted);
    k_embed     <<<(B_ * EMB_) / 256, 256, 0, stream>>>(tokens, emb_tab, embedded);
    k_gx        <<<dim3(12, 8), 256, 0, stream>>>(embedded, weighted, W_ih, b_ih, gx);
    k_gh        <<<dim3(12, 8), 256, 0, stream>>>(hidden, W_hh, b_hh, gh);
    k_gru       <<<(B_ * DEC_) / 256, 256, 0, stream>>>(gx, gh, hidden, h_new);
    k_logits    <<<V_ / 128, 256, 0, stream>>>(h_new, weighted, embedded, W_out, b_out, logits);
}
